// MixAttention_v2_57492432224656
// MI455X (gfx1250) — matrix operand order repaired and checked by reading, not compiled, not run on hardware
//
#include <hip/hip_runtime.h>
#include <hip/hip_bf16.h>
#include <math.h>

// MixAttention for MI455X (gfx1250, wave32, WMMA).
// B=16 N=4096 M=64 C=192 H=6 D=32. All matmuls via v_wmma_f32_16x16x32_bf16.
// ~19.5 GFLOP over ~100MB HBM traffic -> compute-side; bf16 WMMA everywhere,
// softmax in f32, intermediates bf16 in workspace (~77 MB).
// Fragment loads are vectorized to b128 (per-lane A/B data is two contiguous
// 16-byte runs in the ISA layout); bf16 converts use the native cast.

typedef __attribute__((ext_vector_type(16))) __bf16 v16bf;
typedef __attribute__((ext_vector_type(8)))  __bf16 v8bf;
typedef __attribute__((ext_vector_type(8)))  float  v8f;
typedef __attribute__((ext_vector_type(4)))  float  v4f;

static constexpr int Bb = 16, Nn = 4096, Mm = 64, Cc = 192, Hh = 6, Dd = 32;

#define DEVINL __device__ __forceinline__

DEVINL __bf16 f2bf(float f) { return (__bf16)f; }   // native v_cvt to bf16

DEVINL v8f wmma_bf16(v16bf a, v16bf b, v8f c) {
  // D(16x16,f32) = A(16x32,bf16) * B(32x16,bf16) + C
  return __builtin_amdgcn_wmma_f32_16x16x32_bf16(false, a, false, b, (short)0, c,
                                                 false, false);
}

// ---- fragment loaders (CDNA5 ISA 7.12.2 layouts) -------------------------
// A (16-bit, 16x32 MxK): lane holds row M=lane&15, koff=(lane>>4)*8;
// elements are runs k=koff+0..7 and k=16+koff+0..7  -> two b128 loads.
DEVINL v16bf load_a(const __bf16* p, int ld) {
  int lane = threadIdx.x & 31;
  const __bf16* r = p + (size_t)(lane & 15) * ld + (lane >> 4) * 8;
  v8bf lo = *(const v8bf*)(r);
  v8bf hi = *(const v8bf*)(r + 16);
  return __builtin_shufflevector(lo, hi, 0, 1, 2, 3, 4, 5, 6, 7,
                                 8, 9, 10, 11, 12, 13, 14, 15);
}

DEVINL v16bf load_a_f32(const float* p, int ld) {   // fp32 source, convert
  int lane = threadIdx.x & 31;
  const float* r = p + (size_t)(lane & 15) * ld + (lane >> 4) * 8;
  v4f c0 = *(const v4f*)(r);
  v4f c1 = *(const v4f*)(r + 4);
  v4f c2 = *(const v4f*)(r + 16);
  v4f c3 = *(const v4f*)(r + 20);
  v16bf a;
#pragma unroll
  for (int i = 0; i < 4; ++i) {
    a[i]      = f2bf(c0[i]);
    a[4 + i]  = f2bf(c1[i]);
    a[8 + i]  = f2bf(c2[i]);
    a[12 + i] = f2bf(c3[i]);
  }
  return a;
}

// B (32x16 KxN): lane holds column n=lane&15, koff=(lane>>4)*8; k=koff+0..7, 16+koff+0..7.
// Source stored row-major [N][K] (B transposed, e.g. weight rows): contiguous.
DEVINL v16bf load_b_nk(const __bf16* p, int ld) {
  int lane = threadIdx.x & 31;
  const __bf16* r = p + (size_t)(lane & 15) * ld + (lane >> 4) * 8;
  v8bf lo = *(const v8bf*)(r);
  v8bf hi = *(const v8bf*)(r + 16);
  return __builtin_shufflevector(lo, hi, 0, 1, 2, 3, 4, 5, 6, 7,
                                 8, 9, 10, 11, 12, 13, 14, 15);
}

// Source stored row-major [K][N]: column gather (strided).
DEVINL v16bf load_b_kn(const __bf16* p, int ld) {
  int lane = threadIdx.x & 31;
  int n = lane & 15, kb = (lane >> 4) * 8;
  v16bf b;
#pragma unroll
  for (int i = 0; i < 16; ++i) b[i] = p[(size_t)(kb + i + (i >= 8 ? 8 : 0)) * ld + n];
  return b;
}

// ---- kernels -------------------------------------------------------------

__global__ void convert_f32_to_bf16(const float* __restrict__ src,
                                    __bf16* __restrict__ dst, int n) {
  int i = blockIdx.x * blockDim.x + threadIdx.x;
  if (i < n) dst[i] = f2bf(src[i]);
}

// out[R x CC] (bf16) = A[R x K](fp32) * W[CC x K]^T (bf16) + bias
template <int K>
__global__ __launch_bounds__(256) void gemm_f32A_bf16out(
    const float* __restrict__ A, const __bf16* __restrict__ Wb,
    const float* __restrict__ bias, __bf16* __restrict__ out,
    int R, int CCdim) {
  int wid = threadIdx.x >> 5, lane = threadIdx.x & 31;
  int ctiles = CCdim >> 4;
  int tile = blockIdx.x * (blockDim.x >> 5) + wid;
  if (tile >= (R >> 4) * ctiles) return;
  int rt = tile / ctiles, cc = tile % ctiles;
  const float*  Ap = A  + (size_t)(rt * 16) * K;
  const __bf16* Wp = Wb + (size_t)(cc * 16) * K;
  v8f acc = {};
#pragma unroll
  for (int ks = 0; ks < K; ks += 32)
    acc = wmma_bf16(load_a_f32(Ap + ks, K), load_b_nk(Wp + ks, K), acc);
  int col = cc * 16 + (lane & 15);
  int rbase = rt * 16 + 8 * (lane >> 4);
  float bi = bias[col];
#pragma unroll
  for (int v = 0; v < 8; ++v)
    out[(size_t)(rbase + v) * CCdim + col] = f2bf(acc[v] + bi);
}

// out[R x CC] (fp32) = A[R x K](bf16) * W[CC x K]^T (bf16) + bias
template <int K>
__global__ __launch_bounds__(256) void gemm_bf16A_f32out(
    const __bf16* __restrict__ A, const __bf16* __restrict__ Wb,
    const float* __restrict__ bias, float* __restrict__ out,
    int R, int CCdim) {
  int wid = threadIdx.x >> 5, lane = threadIdx.x & 31;
  int ctiles = CCdim >> 4;
  int tile = blockIdx.x * (blockDim.x >> 5) + wid;
  if (tile >= (R >> 4) * ctiles) return;
  int rt = tile / ctiles, cc = tile % ctiles;
  const __bf16* Ap = A  + (size_t)(rt * 16) * K;
  const __bf16* Wp = Wb + (size_t)(cc * 16) * K;
  v8f acc = {};
#pragma unroll
  for (int ks = 0; ks < K; ks += 32)
    acc = wmma_bf16(load_a(Ap + ks, K), load_b_nk(Wp + ks, K), acc);
  int col = cc * 16 + (lane & 15);
  int rbase = rt * 16 + 8 * (lane >> 4);
  float bi = bias[col];
#pragma unroll
  for (int v = 0; v < 8; ++v)
    out[(size_t)(rbase + v) * CCdim + col] = acc[v] + bi;
}

// attn_x: one wave per (b, h, 16-row tile of N). scores = q k^T (16x64),
// row softmax over M=64, out = P v2 (16x32). qv1/kv2 are [rows][384] bf16.
__global__ __launch_bounds__(256) void attn_x_kernel(
    const __bf16* __restrict__ qv1, const __bf16* __restrict__ kv2,
    __bf16* __restrict__ ox, float scale_x) {
  __shared__ __align__(16) __bf16 pbuf[8][16 * 64];  // per-wave P staging
  int wid = threadIdx.x >> 5, lane = threadIdx.x & 31;
  int gw = blockIdx.x * 8 + wid;
  int nt = gw & 255;
  int bh = gw >> 8;
  int h = bh % Hh, b = bh / Hh;
  int colb = lane & 15, rowb = 8 * (lane >> 4);

  v16bf aq = load_a(qv1 + ((size_t)(b * Nn + nt * 16)) * 384 + h * 32, 384);
  v8f s[4];
#pragma unroll
  for (int mt = 0; mt < 4; ++mt) {
    v16bf bk = load_b_nk(kv2 + ((size_t)(b * Mm + mt * 16)) * 384 + h * 32, 384);
    v8f z = {};
    s[mt] = wmma_bf16(aq, bk, z);
  }
  // row softmax (rows live on fixed VGPR within a 16-lane half)
  float rmax[8], rsum[8];
#pragma unroll
  for (int v = 0; v < 8; ++v) {
    float m = s[0][v] * scale_x;
#pragma unroll
    for (int mt = 1; mt < 4; ++mt) m = fmaxf(m, s[mt][v] * scale_x);
    for (int off = 1; off < 16; off <<= 1) m = fmaxf(m, __shfl_xor(m, off, 32));
    rmax[v] = m;
  }
#pragma unroll
  for (int v = 0; v < 8; ++v) {
    float t = 0.f;
#pragma unroll
    for (int mt = 0; mt < 4; ++mt) {
      float e = __expf(s[mt][v] * scale_x - rmax[v]);
      s[mt][v] = e;
      t += e;
    }
    for (int off = 1; off < 16; off <<= 1) t += __shfl_xor(t, off, 32);
    rsum[v] = t;
  }
  // stage normalized P (16x64) to LDS for re-fragmenting as A
#pragma unroll
  for (int mt = 0; mt < 4; ++mt)
#pragma unroll
    for (int v = 0; v < 8; ++v)
      pbuf[wid][(rowb + v) * 64 + mt * 16 + colb] = f2bf(s[mt][v] / rsum[v]);
  asm volatile("s_wait_dscnt 0" ::: "memory");

#pragma unroll
  for (int dt = 0; dt < 2; ++dt) {
    v8f acc = {};
#pragma unroll
    for (int ks = 0; ks < 2; ++ks) {
      v16bf ap = load_a(&pbuf[wid][ks * 32], 64);
      const __bf16* vp =
          kv2 + ((size_t)(b * Mm + ks * 32)) * 384 + Cc + h * 32 + dt * 16;
      acc = wmma_bf16(ap, load_b_kn(vp, 384), acc);
    }
#pragma unroll
    for (int v = 0; v < 8; ++v)
      ox[(size_t)(b * Nn + nt * 16 + rowb + v) * Cc + h * 32 + dt * 16 + colb] =
          f2bf(acc[v]);
  }
}

// attn_c: one workgroup (4 waves) per (b,h). Column softmax over N=4096,
// out_c = softmax(scores^T * scale) @ v1 (64x32). Two-pass (max, then
// exp/sum/accumulate) with score recompute via WMMA.
__global__ __launch_bounds__(128) void attn_c_kernel(
    const __bf16* __restrict__ qv1, const __bf16* __restrict__ kv2,
    __bf16* __restrict__ oc, float scale_c) {
  __shared__ float s_wmax[4 * 64];
  __shared__ float s_colmax[64];
  __shared__ float s_wsum[4 * 64];
  __shared__ float s_accP[4 * 64 * 32];
  __shared__ __align__(16) __bf16 s_pT[4][64 * 32];

  int wid = threadIdx.x >> 5, lane = threadIdx.x & 31;
  int h = blockIdx.x % Hh, b = blockIdx.x / Hh;
  int colb = lane & 15, rowb = 8 * (lane >> 4);

  v16bf bk[4];
#pragma unroll
  for (int mt = 0; mt < 4; ++mt)
    bk[mt] = load_b_nk(kv2 + ((size_t)(b * Mm + mt * 16)) * 384 + h * 32, 384);

  // phase 1: per-column max over all N
  float vmax[4] = {-1e30f, -1e30f, -1e30f, -1e30f};
  for (int nt = wid; nt < Nn / 16; nt += 4) {
    const __bf16* qp = qv1 + ((size_t)(b * Nn + nt * 16)) * 384 + h * 32;
    __builtin_prefetch(qp + (size_t)4 * 16 * 384, 0, 3);  // next tile for wave
    v16bf aq = load_a(qp, 384);
#pragma unroll
    for (int mt = 0; mt < 4; ++mt) {
      v8f z = {};
      v8f sc = wmma_bf16(aq, bk[mt], z);
      float m = sc[0];
#pragma unroll
      for (int v = 1; v < 8; ++v) m = fmaxf(m, sc[v]);
      vmax[mt] = fmaxf(vmax[mt], m);
    }
  }
#pragma unroll
  for (int mt = 0; mt < 4; ++mt)
    vmax[mt] = fmaxf(vmax[mt], __shfl_xor(vmax[mt], 16, 32));
  if (lane < 16)
#pragma unroll
    for (int mt = 0; mt < 4; ++mt) s_wmax[wid * 64 + mt * 16 + lane] = vmax[mt];
  __syncthreads();
  if (threadIdx.x < 64) {
    float m = s_wmax[threadIdx.x];
#pragma unroll
    for (int w = 1; w < 4; ++w) m = fmaxf(m, s_wmax[w * 64 + threadIdx.x]);
    s_colmax[threadIdx.x] = m;
  }
  __syncthreads();

  // phase 2: exp, column sums, and P^T @ v1 accumulation
  v8f acc[4][2];
#pragma unroll
  for (int mt = 0; mt < 4; ++mt)
#pragma unroll
    for (int dt = 0; dt < 2; ++dt) { v8f z = {}; acc[mt][dt] = z; }
  float csum[4] = {0.f, 0.f, 0.f, 0.f};

  for (int nb = wid; nb < Nn / 32; nb += 4) {
    const __bf16* qbase = qv1 + ((size_t)(b * Nn + nb * 32)) * 384 + h * 32;
    __builtin_prefetch(qbase + (size_t)4 * 32 * 384, 0, 3);
#pragma unroll
    for (int sub = 0; sub < 2; ++sub) {
      v16bf aq = load_a(qbase + (size_t)sub * 16 * 384, 384);
#pragma unroll
      for (int mt = 0; mt < 4; ++mt) {
        v8f z = {};
        v8f sc = wmma_bf16(aq, bk[mt], z);
        float cmax = s_colmax[mt * 16 + colb];
#pragma unroll
        for (int v = 0; v < 8; ++v) {
          float e = __expf((sc[v] - cmax) * scale_c);
          csum[mt] += e;
          // transpose into [m][n] layout for the A operand
          s_pT[wid][(mt * 16 + colb) * 32 + sub * 16 + rowb + v] = f2bf(e);
        }
      }
    }
    asm volatile("s_wait_dscnt 0" ::: "memory");
    // v1 B fragments depend only on (nb, dt): load once, reuse across mt
    v16bf bv[2];
#pragma unroll
    for (int dt = 0; dt < 2; ++dt)
      bv[dt] = load_b_kn(qbase - h * 32 + Cc + h * 32 + dt * 16, 384);
#pragma unroll
    for (int mt = 0; mt < 4; ++mt) {
      v16bf ap = load_a(&s_pT[wid][(mt * 16) * 32], 32);
#pragma unroll
      for (int dt = 0; dt < 2; ++dt)
        acc[mt][dt] = wmma_bf16(ap, bv[dt], acc[mt][dt]);
    }
  }
#pragma unroll
  for (int mt = 0; mt < 4; ++mt) csum[mt] += __shfl_xor(csum[mt], 16, 32);
  if (lane < 16)
#pragma unroll
    for (int mt = 0; mt < 4; ++mt) s_wsum[wid * 64 + mt * 16 + lane] = csum[mt];
#pragma unroll
  for (int mt = 0; mt < 4; ++mt)
#pragma unroll
    for (int dt = 0; dt < 2; ++dt)
#pragma unroll
      for (int v = 0; v < 8; ++v)
        s_accP[wid * 2048 + (mt * 16 + rowb + v) * 32 + dt * 16 + colb] =
            acc[mt][dt][v];
  __syncthreads();

  for (int idx = threadIdx.x; idx < 64 * 32; idx += 128) {
    int m = idx >> 5, d = idx & 31;
    float a0 = s_accP[idx] + s_accP[2048 + idx] + s_accP[4096 + idx] +
               s_accP[6144 + idx];
    float s0 = s_wsum[m] + s_wsum[64 + m] + s_wsum[128 + m] + s_wsum[192 + m];
    oc[(size_t)(b * Mm + m) * Cc + h * 32 + d] = f2bf(a0 / s0);
  }
}

// ---- host ----------------------------------------------------------------

extern "C" void kernel_launch(void* const* d_in, const int* in_sizes, int n_in,
                              void* d_out, int out_size, void* d_ws, size_t ws_size,
                              hipStream_t stream) {
  (void)in_sizes; (void)n_in; (void)out_size; (void)ws_size;
  const float* x    = (const float*)d_in[0];
  const float* c    = (const float*)d_in[1];
  const float* Wqv1 = (const float*)d_in[2];
  const float* bqv1 = (const float*)d_in[3];
  const float* Wkv2 = (const float*)d_in[4];
  const float* bkv2 = (const float*)d_in[5];
  const float* Wpx  = (const float*)d_in[6];
  const float* bpx  = (const float*)d_in[7];
  const float* Wpc  = (const float*)d_in[8];
  const float* bpc  = (const float*)d_in[9];

  float* out_x = (float*)d_out;
  float* out_c = out_x + (size_t)Bb * Nn * Cc;

  // workspace carve (~77 MB total)
  char* p = (char*)d_ws;
  auto carve = [&](size_t bytes) {
    char* r = p;
    p += (bytes + 255) & ~(size_t)255;
    return r;
  };
  __bf16* qv1b  = (__bf16*)carve((size_t)Bb * Nn * 384 * 2);  // 50.3 MB
  __bf16* kv2b  = (__bf16*)carve((size_t)Bb * Mm * 384 * 2);  // 0.8 MB
  __bf16* oxb   = (__bf16*)carve((size_t)Bb * Nn * Cc * 2);   // 25.2 MB
  __bf16* ocb   = (__bf16*)carve((size_t)Bb * Mm * Cc * 2);   // 0.4 MB
  __bf16* Wqv1b = (__bf16*)carve((size_t)2 * Cc * Cc * 2);
  __bf16* Wkv2b = (__bf16*)carve((size_t)2 * Cc * Cc * 2);
  __bf16* Wpxb  = (__bf16*)carve((size_t)Cc * Cc * 2);
  __bf16* Wpcb  = (__bf16*)carve((size_t)Cc * Cc * 2);

  const float scale   = 1.0f / sqrtf((float)Cc);
  const float scale_x = 0.5f * scale;   // log_4096(64) = 0.5 exactly
  const float scale_c = scale;

  convert_f32_to_bf16<<<(2 * Cc * Cc + 255) / 256, 256, 0, stream>>>(Wqv1, Wqv1b, 2 * Cc * Cc);
  convert_f32_to_bf16<<<(2 * Cc * Cc + 255) / 256, 256, 0, stream>>>(Wkv2, Wkv2b, 2 * Cc * Cc);
  convert_f32_to_bf16<<<(Cc * Cc + 255) / 256, 256, 0, stream>>>(Wpx, Wpxb, Cc * Cc);
  convert_f32_to_bf16<<<(Cc * Cc + 255) / 256, 256, 0, stream>>>(Wpc, Wpcb, Cc * Cc);

  // qv1 = x @ Wqv1^T + b : [65536 x 384], K=192
  {
    int tiles = (Bb * Nn / 16) * (2 * Cc / 16);
    gemm_f32A_bf16out<Cc><<<tiles / 8, 256, 0, stream>>>(x, Wqv1b, bqv1, qv1b,
                                                         Bb * Nn, 2 * Cc);
  }
  // kv2 = c @ Wkv2^T + b : [1024 x 384], K=192
  {
    int tiles = (Bb * Mm / 16) * (2 * Cc / 16);
    gemm_f32A_bf16out<Cc><<<tiles / 8, 256, 0, stream>>>(c, Wkv2b, bkv2, kv2b,
                                                         Bb * Mm, 2 * Cc);
  }
  // attention, x-direction: 16*6*256 = 24576 waves
  attn_x_kernel<<<(Bb * Hh * (Nn / 16)) / 8, 256, 0, stream>>>(qv1b, kv2b, oxb,
                                                               scale_x);
  // attention, c-direction: one WG per (b,h)
  attn_c_kernel<<<Bb * Hh, 128, 0, stream>>>(qv1b, kv2b, ocb, scale_c);

  // output projections (fp32 + bias into d_out)
  {
    int tiles = (Bb * Nn / 16) * (Cc / 16);
    gemm_bf16A_f32out<Cc><<<tiles / 8, 256, 0, stream>>>(oxb, Wpxb, bpx, out_x,
                                                         Bb * Nn, Cc);
  }
  {
    int tiles = (Bb * Mm / 16) * (Cc / 16);
    gemm_bf16A_f32out<Cc><<<tiles / 8, 256, 0, stream>>>(ocb, Wpcb, bpc, out_c,
                                                         Bb * Mm, Cc);
  }
}
